// GNNRegressor_77171972374887
// MI455X (gfx1250) — matrix operand order repaired and checked by reading, not compiled, not run on hardware
//
#include <hip/hip_runtime.h>

#define N_NODES 50000
#define DCH     128
#define N_EDGES 800000
#define NTILES  (N_NODES / 16)   // 3125, exact

typedef __attribute__((ext_vector_type(16))) __bf16 v16bf;
typedef __attribute__((ext_vector_type(8)))  __bf16 v8bf;
typedef __attribute__((ext_vector_type(4)))  __bf16 v4bf;
typedef __attribute__((ext_vector_type(8)))  float  v8f;

// ---- A-fragment builders (16x32 bf16 A layout, lane = m + 16*hi) ----------
// halves[0..7]  = K = base + 16*hi + i       (contiguous 8)
// halves[8..15] = K = base + 16*hi + 8 + i   (contiguous 8)
__device__ __forceinline__ v16bf frag_from_f32(const float* __restrict__ p, float s) {
  const float4* q = (const float4*)p;
  float4 f0 = q[0], f1 = q[1], f2 = q[2], f3 = q[3];
  v16bf a;
  a[0]=(__bf16)(f0.x*s); a[1]=(__bf16)(f0.y*s); a[2]=(__bf16)(f0.z*s); a[3]=(__bf16)(f0.w*s);
  a[4]=(__bf16)(f1.x*s); a[5]=(__bf16)(f1.y*s); a[6]=(__bf16)(f1.z*s); a[7]=(__bf16)(f1.w*s);
  a[8]=(__bf16)(f2.x*s); a[9]=(__bf16)(f2.y*s); a[10]=(__bf16)(f2.z*s); a[11]=(__bf16)(f2.w*s);
  a[12]=(__bf16)(f3.x*s); a[13]=(__bf16)(f3.y*s); a[14]=(__bf16)(f3.z*s); a[15]=(__bf16)(f3.w*s);
  return a;
}
__device__ __forceinline__ v16bf load_a(const float* __restrict__ row, int off, float s) {
  return frag_from_f32(row + off, s);
}
__device__ __forceinline__ v16bf load_a(const __bf16* __restrict__ row, int off, float) {
  v8bf lo = *(const v8bf*)(row + off);
  v8bf hi = *(const v8bf*)(row + off + 8);
  v16bf a;
#pragma unroll
  for (int i = 0; i < 8; ++i) { a[i] = lo[i]; a[i + 8] = hi[i]; }
  return a;
}

// ---- degree count ---------------------------------------------------------
__global__ void degree_count(const int* __restrict__ dst, float* __restrict__ cnt) {
  int e = blockIdx.x * blockDim.x + threadIdx.x;
  if (e < N_EDGES) atomicAdd(&cnt[dst[e]], 1.0f);
}

// ---- scatter-add of messages: 32 lanes per edge, 4 channels per lane ------
__global__ void scatter_f32(const float* __restrict__ x, const int* __restrict__ src,
                            const int* __restrict__ dst, float* __restrict__ sums) {
  int gid = blockIdx.x * blockDim.x + threadIdx.x;
  int e = gid >> 5;
  if (e >= N_EDGES) return;
  int c = (gid & 31) * 4;
  int s = src[e], d = dst[e];
  float4 v = *(const float4*)(x + s * DCH + c);
  float* q = sums + d * DCH + c;
  atomicAdd(q + 0, v.x); atomicAdd(q + 1, v.y);
  atomicAdd(q + 2, v.z); atomicAdd(q + 3, v.w);
}

__global__ void scatter_bf16(const __bf16* __restrict__ h, const int* __restrict__ src,
                             const int* __restrict__ dst, float* __restrict__ sums) {
  int gid = blockIdx.x * blockDim.x + threadIdx.x;
  int e = gid >> 5;
  if (e >= N_EDGES) return;
  int c = (gid & 31) * 4;
  int s = src[e], d = dst[e];
  v4bf v = *(const v4bf*)(h + s * DCH + c);
  float* q = sums + d * DCH + c;
  atomicAdd(q + 0, (float)v[0]); atomicAdd(q + 1, (float)v[1]);
  atomicAdd(q + 2, (float)v[2]); atomicAdd(q + 3, (float)v[3]);
}

// ---- fused SAGE layer: out = relu( (sums/max(cnt,1))@Wl + bl + xin@Wr ) ---
// One wave per 16-node tile; weights bf16-transposed in LDS (WT[n][k]).
template <typename TIn>
__global__ __launch_bounds__(256)
void sage_gemm(const float* __restrict__ sums, const float* __restrict__ cnt,
               const TIn* __restrict__ xin,
               const float* __restrict__ Wl, const float* __restrict__ bl,
               const float* __restrict__ Wr,
               __bf16* __restrict__ out) {
  __shared__ __bf16 WTl[DCH * DCH];
  __shared__ __bf16 WTr[DCH * DCH];
  for (int idx = threadIdx.x; idx < DCH * DCH; idx += 256) {
    int k = idx >> 7, n = idx & 127;
    WTl[n * DCH + k] = (__bf16)Wl[idx];
    WTr[n * DCH + k] = (__bf16)Wr[idx];
  }
  __syncthreads();

  int wave = threadIdx.x >> 5;
  int lane = threadIdx.x & 31;
  int tile = blockIdx.x * 8 + wave;
  if (tile >= NTILES) return;           // wave-uniform: EXEC all-1s for WMMA

  int m   = lane & 15;                  // A row within tile / C-D column
  int hi  = lane >> 4;
  int row = tile * 16 + m;

  float invc = 1.0f / fmaxf(cnt[row], 1.0f);

  v16bf a_agg[4], a_x[4];
#pragma unroll
  for (int kb = 0; kb < 4; ++kb) {
    int off = 32 * kb + 16 * hi;
    a_agg[kb] = frag_from_f32(sums + row * DCH + off, invc);
    a_x[kb]   = load_a(xin + row * DCH, off, 1.0f);
  }

#pragma unroll
  for (int nt = 0; nt < 8; ++nt) {
    int ncol = nt * 16 + m;
    v8f acc = {};
#pragma unroll
    for (int kb = 0; kb < 4; ++kb) {
      // B-fragment: halves[i] = W[32*kb + 16*hi + i][ncol], contiguous in WT
      v16bf bL = *(const v16bf*)(&WTl[ncol * DCH + 32 * kb + 16 * hi]);
      acc = __builtin_amdgcn_wmma_f32_16x16x32_bf16(false, a_agg[kb], false, bL,
                                                    (short)0, acc, false, false);
      v16bf bR = *(const v16bf*)(&WTr[ncol * DCH + 32 * kb + 16 * hi]);
      acc = __builtin_amdgcn_wmma_f32_16x16x32_bf16(false, a_x[kb], false, bR,
                                                    (short)0, acc, false, false);
    }
    float bias = bl[ncol];
#pragma unroll
    for (int r = 0; r < 8; ++r) {
      float v = acc[r] + bias;
      v = v > 0.0f ? v : 0.0f;          // ReLU
      out[(tile * 16 + r + 8 * hi) * DCH + ncol] = (__bf16)v;
    }
  }
}

// ---- final head: one wave per node, dot(h2[n], Wout) + bout ---------------
__global__ void out_head(const __bf16* __restrict__ h2, const float* __restrict__ Wout,
                         const float* __restrict__ bout, float* __restrict__ out) {
  int gid = blockIdx.x * blockDim.x + threadIdx.x;
  int n = gid >> 5;
  if (n >= N_NODES) return;
  int c = (gid & 31) * 4;
  v4bf v = *(const v4bf*)(h2 + n * DCH + c);
  float s = (float)v[0] * Wout[c] + (float)v[1] * Wout[c + 1] +
            (float)v[2] * Wout[c + 2] + (float)v[3] * Wout[c + 3];
#pragma unroll
  for (int off = 16; off > 0; off >>= 1) s += __shfl_xor(s, off, 32);
  if ((gid & 31) == 0) out[n] = s + bout[0];
}

extern "C" void kernel_launch(void* const* d_in, const int* in_sizes, int n_in,
                              void* d_out, int out_size, void* d_ws, size_t ws_size,
                              hipStream_t stream) {
  const float* x    = (const float*)d_in[0];
  const int*   ei   = (const int*)d_in[1];       // (2, E) int32
  const int*   src  = ei;
  const int*   dst  = ei + N_EDGES;
  const float* W1l  = (const float*)d_in[2];
  const float* b1l  = (const float*)d_in[3];
  const float* W1r  = (const float*)d_in[4];
  const float* W2l  = (const float*)d_in[5];
  const float* b2l  = (const float*)d_in[6];
  const float* W2r  = (const float*)d_in[7];
  const float* Wout = (const float*)d_in[8];
  const float* bout = (const float*)d_in[9];

  char* ws = (char*)d_ws;
  float*  sums = (float*)ws;                                     // 25,600,000 B
  float*  cnt  = (float*)(ws + 25600000);                        //    200,000 B
  __bf16* h1   = (__bf16*)(ws + 25800000);                       // 12,800,000 B
  __bf16* h2   = (__bf16*)(ws + 38600000);                       // 12,800,000 B

  const int scatterBlocks = (N_EDGES * 32) / 256;                // 100,000
  const int gemmBlocks    = (NTILES + 7) / 8;                    // 391

  // Layer 1
  hipMemsetAsync(sums, 0, (size_t)N_NODES * DCH * sizeof(float), stream);
  hipMemsetAsync(cnt, 0, (size_t)N_NODES * sizeof(float), stream);
  degree_count<<<(N_EDGES + 255) / 256, 256, 0, stream>>>(dst, cnt);
  scatter_f32<<<scatterBlocks, 256, 0, stream>>>(x, src, dst, sums);
  sage_gemm<float><<<gemmBlocks, 256, 0, stream>>>(sums, cnt, x, W1l, b1l, W1r, h1);

  // Layer 2
  hipMemsetAsync(sums, 0, (size_t)N_NODES * DCH * sizeof(float), stream);
  scatter_bf16<<<scatterBlocks, 256, 0, stream>>>(h1, src, dst, sums);
  sage_gemm<__bf16><<<gemmBlocks, 256, 0, stream>>>(sums, cnt, h1, W2l, b2l, W2r, h2);

  // Head
  out_head<<<(N_NODES * 32 + 255) / 256, 256, 0, stream>>>(h2, Wout, bout, (float*)d_out);
}
